// GriffinGate_21045339750518
// MI455X (gfx1250) — matrix operand order repaired and checked by reading, not compiled, not run on hardware
//
#include <hip/hip_runtime.h>
#include <hip/hip_bf16.h>

typedef __attribute__((ext_vector_type(16))) __bf16 v16bf;
typedef __attribute__((ext_vector_type(8)))  __bf16 v8bf;
typedef __attribute__((ext_vector_type(4)))  __bf16 v4bf;
typedef __attribute__((ext_vector_type(8)))  float  v8f;
typedef __attribute__((ext_vector_type(4)))  float  v4f;

#define WIDTH      2048
#define NBLK       16
#define BLK        128
#define ROWS_TOTAL (4 * 8192)
#define TILE_M     16
#define LDS_STRIDE (WIDTH + 16)   // pad 32B so 16 rows hit different LDS banks
#define WP_ELEMS   (NBLK * BLK * BLK)   // 262144 packed bf16 weights

// ---------------------------------------------------------------------------
// Kernel 1: repack w [16,128,128] f32 -> bf16 in d_ws, pre-swizzled into the
// per-lane WMMA B-fragment layout for v_wmma_f32_16x16x32_bf16:
//   fragment f = ((h*8 + nt)*4 + ks); per lane: 16 contiguous bf16 (32 B)
//   lane L: n = nt*16 + (L%16); element e: k = ks*32 + (L<16 ? 0 : 16) + e
// ---------------------------------------------------------------------------
__global__ __launch_bounds__(256)
void pack_w_bf16(const float* __restrict__ w, __bf16* __restrict__ wp) {
  int idx  = blockIdx.x * 256 + threadIdx.x;     // 16*8*4*32*16 = 262144 total
  int e    = idx & 15;
  int lane = (idx >> 4) & 31;
  int ks   = (idx >> 9) & 3;
  int nt   = (idx >> 11) & 7;
  int h    = idx >> 14;
  int k = ks * 32 + ((lane < 16) ? 0 : 16) + e;
  int n = nt * 16 + (lane & 15);
  wp[idx] = (__bf16)w[h * BLK * BLK + k * BLK + n];
}

// ---------------------------------------------------------------------------
// Kernel 1b: precompute nsp8[col] = -C_VALUE * softplus(a_param[col]).
// Only 2048 values for the whole problem -- hoists the expensive log1pf
// expansion out of the 67M-element hot kernel.
// ---------------------------------------------------------------------------
__global__ __launch_bounds__(256)
void prep_gate(const float* __restrict__ a_param, float* __restrict__ nsp8) {
  int c = blockIdx.x * 256 + threadIdx.x;        // 2048 total
  float a0 = a_param[c];
  float sp = fmaxf(a0, 0.0f) + log1pf(__expf(-fabsf(a0)));
  nsp8[c] = -8.0f * sp;                          // C_VALUE = 8
}

// ---------------------------------------------------------------------------
// Kernel 2: fused LayerNorm + block-diagonal GEMM (bf16 WMMA) + gate.
// 16 waves / workgroup; workgroup owns a 16-row tile of x.
//   Phase 1: wave r LayerNorms row r  -> bf16 tile in LDS
//   Phase 2: wave h computes the 16x128 output of diagonal block h via
//            8 N-tiles x 4 chained v_wmma_f32_16x16x32_bf16, then applies
//            the gate (fast-rcp sigmoid, precomputed softplus) and writes
//            alpha / beta.
// ---------------------------------------------------------------------------
__global__ __launch_bounds__(512)
void griffin_gate(const float*  __restrict__ x,
                  const float*  __restrict__ lnw,
                  const float*  __restrict__ lnb,
                  const __bf16* __restrict__ wp,
                  const float*  __restrict__ bias_p,
                  const float*  __restrict__ nsp8_p,
                  float*        __restrict__ out) {
  __shared__ unsigned short sA[TILE_M * LDS_STRIDE];

  const int  tid  = threadIdx.x;
  const int  wave = tid >> 5;
  const int  lane = tid & 31;
  const long rowBase = (long)blockIdx.x * TILE_M;

  // ---------- Phase 1: LayerNorm (one wave per row) ----------
  {
    const int r = wave;
    const float* xr = x + (rowBase + r) * WIDTH;
    v4f  v[16];
    float s = 0.f, s2 = 0.f;
#pragma unroll
    for (int i = 0; i < 16; ++i) {
      v[i] = *(const v4f*)(xr + 4 * (i * 32 + lane));
#pragma unroll
      for (int j = 0; j < 4; ++j) { s += v[i][j]; s2 += v[i][j] * v[i][j]; }
    }
#pragma unroll
    for (int m = 16; m >= 1; m >>= 1) {   // wave32 butterfly reduction
      s  += __shfl_xor(s,  m, 32);
      s2 += __shfl_xor(s2, m, 32);
    }
    const float mu  = s * (1.0f / WIDTH);
    const float var = s2 * (1.0f / WIDTH) - mu * mu;
    const float rs  = rsqrtf(var + 1e-5f);

    unsigned short* dst = sA + r * LDS_STRIDE;
#pragma unroll
    for (int i = 0; i < 16; ++i) {
      const int c0 = 4 * (i * 32 + lane);
      v4f wv = *(const v4f*)(lnw + c0);
      v4f bv = *(const v4f*)(lnb + c0);
      v4bf o;
#pragma unroll
      for (int j = 0; j < 4; ++j)
        o[j] = (__bf16)((v[i][j] - mu) * rs * wv[j] + bv[j]);
      *(v4bf*)(dst + c0) = o;               // ds_store_b64
    }
  }
  __syncthreads();

  // ---------- Phase 2: WMMA GEMM for diagonal block h = wave ----------
  const int h  = wave;
  const int m  = lane & 15;
  const int kb = (lane < 16) ? 0 : 16;      // A-fragment K base (contiguous, matches packed B)

  // Load the 4 A fragments (16x32 bf16 each) once; reuse for all 8 N-tiles.
  v16bf afrag[4];
#pragma unroll
  for (int ks = 0; ks < 4; ++ks) {
    const unsigned short* ab = sA + m * LDS_STRIDE + h * BLK + ks * 32 + kb;
    v8bf lo = *(const v8bf*)(ab);           // K = kb .. kb+7      (ds_load_b128)
    v8bf hi = *(const v8bf*)(ab + 8);       // K = kb+8 .. kb+15   (ds_load_b128)
    v16bf a;
#pragma unroll
    for (int j = 0; j < 8; ++j) { a[j] = lo[j]; a[j + 8] = hi[j]; }
    afrag[ks] = a;
  }

  const int  n   = lane & 15;
  const long TOT = (long)ROWS_TOTAL * WIDTH;
  const v16bf* Bbase = (const v16bf*)wp;

  for (int nt = 0; nt < 8; ++nt) {
    const v16bf* Bp = Bbase + (long)((h * 8 + nt) * 4) * 32 + lane;
    if (nt < 7)
      __builtin_prefetch((const void*)(Bp + 4 * 32), 0, 1);  // global_prefetch_b8

    v8f acc = {};
#pragma unroll
    for (int ks = 0; ks < 4; ++ks) {
      v16bf bfrag = Bp[ks * 32];
      acc = __builtin_amdgcn_wmma_f32_16x16x32_bf16(
          /*neg_a=*/false, afrag[ks], /*neg_b=*/false, bfrag,
          /*c_mod=*/(short)0, acc, /*reuse_a=*/false, /*reuse_b=*/false);
    }

    // ---- gate epilogue on the accumulator fragment ----
    const int   colg = h * BLK + nt * 16 + n;
    const float bias = bias_p[colg];
    const float nsp8 = nsp8_p[colg];        // precomputed -8*softplus(a_param)
#pragma unroll
    for (int r2 = 0; r2 < 8; ++r2) {
      const long  row = rowBase + r2 + ((lane < 16) ? 0 : 8);
      const float g   = acc[r2] + bias;
      // sigmoid via hw v_rcp_f32 (alpha is clipped to [0.05,0.95] anyway)
      const float t   = __expf(-g);                      // D_VALUE = 1
      const float sig = __builtin_amdgcn_rcpf(1.0f + t);
      float alpha = __expf(nsp8 * sig);
      alpha = fminf(fmaxf(alpha, 0.05f), 0.95f);
      out[row * WIDTH + colg]       = alpha;
      out[TOT + row * WIDTH + colg] = 1.0f - alpha;
    }
  }
}

// ---------------------------------------------------------------------------
// Inputs (setup_inputs order): x, ln_weight, ln_bias, w, b, a_param (all f32)
// Output: alpha||beta flat f32 (2 * 4*8192*2048 elements)
// Workspace: [0, 512KB) packed bf16 weights; [512KB, 512KB+8KB) nsp8 table
// ---------------------------------------------------------------------------
extern "C" void kernel_launch(void* const* d_in, const int* in_sizes, int n_in,
                              void* d_out, int out_size, void* d_ws, size_t ws_size,
                              hipStream_t stream) {
  const float* x   = (const float*)d_in[0];
  const float* lnw = (const float*)d_in[1];
  const float* lnb = (const float*)d_in[2];
  const float* w   = (const float*)d_in[3];
  const float* b   = (const float*)d_in[4];
  const float* ap  = (const float*)d_in[5];
  float*   out  = (float*)d_out;
  __bf16*  wp   = (__bf16*)d_ws;                        // 512 KB packed weights
  float*   nsp8 = (float*)((char*)d_ws + WP_ELEMS * sizeof(__bf16));

  // 1) pack weights to bf16 WMMA B-fragment layout + precompute gate table
  pack_w_bf16<<<WP_ELEMS / 256, 256, 0, stream>>>(w, wp);
  prep_gate<<<WIDTH / 256, 256, 0, stream>>>(ap, nsp8);

  // 2) fused LN + block-diag WMMA GEMM + gate
  griffin_gate<<<ROWS_TOTAL / TILE_M, 512, 0, stream>>>(
      x, lnw, lnb, wp, b, nsp8, out);
}
